// GAT_34720515621232
// MI455X (gfx1250) — matrix operand order repaired and checked by reading, not compiled, not run on hardware
//
#include <hip/hip_runtime.h>

typedef __attribute__((ext_vector_type(16))) __bf16 v16bf;
typedef __attribute__((ext_vector_type(8)))  float  v8f;

#define NN 6144
#define NEG_INF (-9e15f)
#define LALPHA 0.2f

#if __has_builtin(__builtin_amdgcn_global_load_async_to_lds_b128) && \
    __has_builtin(__builtin_amdgcn_s_wait_asynccnt)
#define HAVE_ASYNC 1
#else
#define HAVE_ASYNC 0
#endif

#if HAVE_ASYNC
typedef int v4i_gcc __attribute__((vector_size(16)));            // matches builtin param
typedef __attribute__((address_space(1))) v4i_gcc gv4i;          // global (device) AS
typedef __attribute__((address_space(3))) v4i_gcc lv4i;          // LDS AS
__device__ __forceinline__ void async_cp16(void* lds_dst, const void* gsrc) {
    __builtin_amdgcn_global_load_async_to_lds_b128((gv4i*)gsrc, (lv4i*)lds_dst, 0, 0);
}
#endif

__device__ __forceinline__ float lrelu(float x) { return fmaxf(x, LALPHA * x); }
__device__ __forceinline__ float eluf(float x)  { return x > 0.f ? x : expm1f(x); }

// A-fragment (16x32 bf16, M=lane&15): K index for vector element idx (0..15)
__device__ __forceinline__ int a_k(int lane, int idx) {
    int v = idx >> 1, e = idx & 1;
    int base = (v < 4) ? (2 * v) : (16 + 2 * (v - 4));
    return base + ((lane >> 4) << 3) + e;
}

union Frag16 {                 // 16 bf16 = two 16-byte chunks
    uint4  u[2];
    v16bf  v;
};

// ---------------------------------------------------------------------------
// C[head][M][FDIM] = X[M][KDIM] @ W[head][KDIM][FDIM]   (bf16 WMMA, f32 accum)
// Hf written row-major [H][N][F]; Hb written FEATURE-MAJOR [H][F][N] so the
// aggregation kernel can ds_load_b128 its B fragments.
// ---------------------------------------------------------------------------
template <int KDIM, int FDIM>
__global__ void gemm_xw_kernel(const float* __restrict__ X,
                               const float* __restrict__ W,
                               float* __restrict__ Hf,
                               __bf16* __restrict__ Hb) {
    constexpr int NT = FDIM / 16;
    constexpr int AS = 40;                    // row stride (bf16), 80B = 16B-aligned
    constexpr int KS = 40;
    __shared__ __align__(16) __bf16 lA[64 * AS];        // [row][k]
    __shared__ __align__(16) __bf16 lBT[FDIM * KS];     // [n][k]  (K-contiguous)

    const int head = blockIdx.y;
    const int m0   = blockIdx.x * 64;
    const int tid  = threadIdx.x;
    const int lane = tid & 31, w = tid >> 5;
    const float* Wh = W + (size_t)head * KDIM * FDIM;

    v8f acc[NT] = {};

    for (int kk = 0; kk < KDIM; kk += 32) {
        for (int idx = tid; idx < 64 * 32; idx += 128) {
            int r = idx >> 5, c = idx & 31;
            lA[r * AS + c] = (__bf16)X[(size_t)(m0 + r) * KDIM + kk + c];
        }
        for (int idx = tid; idx < 32 * FDIM; idx += 128) {
            int r = idx / FDIM, c = idx % FDIM;          // r = k, c = n
            lBT[c * KS + r] = (__bf16)Wh[(size_t)(kk + r) * FDIM + c];
        }
        __syncthreads();

        Frag16 af;
        {
            const __bf16* ap = &lA[(w * 16 + (lane & 15)) * AS + ((lane >> 4) << 3)];
            af.u[0] = *(const uint4*)(ap);       // K klo..klo+7
            af.u[1] = *(const uint4*)(ap + 16);  // K 16+klo..16+klo+7
        }
#pragma unroll
        for (int t = 0; t < NT; ++t) {
            Frag16 bf;
            const __bf16* bp = &lBT[(t * 16 + (lane & 15)) * KS + ((lane >> 4) << 3)];
            bf.u[0] = *(const uint4*)(bp);
            bf.u[1] = *(const uint4*)(bp + 16);
            acc[t] = __builtin_amdgcn_wmma_f32_16x16x32_bf16(
                false, af.v, false, bf.v, (short)0, acc[t], false, false);
        }
        __syncthreads();
    }

#pragma unroll
    for (int t = 0; t < NT; ++t) {
        const int n = t * 16 + (lane & 15);
#pragma unroll
        for (int r = 0; r < 8; ++r) {
            const int mrow = m0 + w * 16 + r + ((lane >> 4) << 3);
            const float v  = acc[t][r];
            Hf[(size_t)head * NN * FDIM + (size_t)mrow * FDIM + n] = v;
            Hb[((size_t)head * FDIM + n) * NN + mrow]              = (__bf16)v;  // transposed
        }
    }
}

// ---------------------------------------------------------------------------
// s1[t] = h_row . a1 ; s2[t] = h_row . a2     (t = head*N + i)
// ---------------------------------------------------------------------------
template <int FDIM>
__global__ void s12_kernel(const float* __restrict__ Hf, const float* __restrict__ A,
                           float* __restrict__ s1, float* __restrict__ s2, int HN) {
    int t = blockIdx.x * blockDim.x + threadIdx.x;
    if (t >= HN) return;
    int h = t / NN;
    const float* row = Hf + (size_t)t * FDIM;
    const float* a1  = A + (size_t)h * 2 * FDIM;
    const float* a2  = a1 + FDIM;
    float r1 = 0.f, r2 = 0.f;
#pragma unroll 8
    for (int f = 0; f < FDIM; ++f) { float v = row[f]; r1 += v * a1[f]; r2 += v * a2[f]; }
    s1[t] = r1; s2[t] = r2;
}

// ---------------------------------------------------------------------------
// Row softmax stats: one wave per row, all HEADS share one adj scan.
// Masked entries use the finite NEG_INF exactly like the reference.
// ---------------------------------------------------------------------------
template <int HEADS>
__global__ void maxz_kernel(const int* __restrict__ adj,
                            const float* __restrict__ s1, const float* __restrict__ s2,
                            float* __restrict__ M, float* __restrict__ Z) {
    const int lane = threadIdx.x & 31;
    const int wid  = threadIdx.x >> 5;
    const int i    = blockIdx.x * (blockDim.x >> 5) + wid;
    if (i >= NN) return;

    float s1i[HEADS], mh[HEADS], zh[HEADS];
#pragma unroll
    for (int h = 0; h < HEADS; ++h) { s1i[h] = s1[h * NN + i]; mh[h] = NEG_INF; zh[h] = 0.f; }

    const int* arow = adj + (size_t)i * NN;
    for (int j = lane; j < NN; j += 32) {
        const int a = arow[j];
#pragma unroll
        for (int h = 0; h < HEADS; ++h) {
            const float e = (a > 0) ? lrelu(s1i[h] + s2[h * NN + j]) : NEG_INF;
            if (e > mh[h]) { zh[h] = zh[h] * __expf(mh[h] - e) + 1.f; mh[h] = e; }
            else           { zh[h] += __expf(e - mh[h]); }
        }
    }
#pragma unroll
    for (int off = 16; off >= 1; off >>= 1) {
#pragma unroll
        for (int h = 0; h < HEADS; ++h) {
            const float mo = __shfl_xor(mh[h], off, 32);
            const float zo = __shfl_xor(zh[h], off, 32);
            const float mn = fmaxf(mh[h], mo);
            zh[h] = zh[h] * __expf(mh[h] - mn) + zo * __expf(mo - mn);
            mh[h] = mn;
        }
    }
    if (lane == 0) {
#pragma unroll
        for (int h = 0; h < HEADS; ++h) { M[h * NN + i] = mh[h]; Z[h * NN + i] = zh[h]; }
    }
}

// ---------------------------------------------------------------------------
// Flash-style masked-softmax aggregation: Out[h][i][f] = elu((P @ H)/Z).
// P built per-lane directly in the WMMA A-fragment layout; B fragments come
// from a K-contiguous (transposed) LDS tile via ds_load_b128 pairs.
// adj + h tiles staged through LDS with async b128 copies when available.
// ---------------------------------------------------------------------------
template <int FDIM, int HEADS>
__global__ void pv_kernel(const int* __restrict__ adj,
                          const float* __restrict__ s1, const float* __restrict__ s2,
                          const float* __restrict__ M, const float* __restrict__ Z,
                          const __bf16* __restrict__ HbT,   // [HEADS][FDIM][N]
                          float* __restrict__ Out) {        // [HEADS][N][FDIM]
    constexpr int NT  = FDIM / 16;
    constexpr int RPB = 16 * (4 / HEADS);   // rows per block
    constexpr int KS  = 40;                 // padded K stride (bf16)
    __shared__ __align__(16) int    lAdj[RPB * 32];
    __shared__ __align__(16) __bf16 lHT[HEADS * FDIM * KS];   // [(h,f)][k]
    __shared__ __align__(16) float  lS2[HEADS * 32];

    const int tid = threadIdx.x, lane = tid & 31, w = tid >> 5;
    const int head = w % HEADS;
    const int rblk = w / HEADS;
    const int i0b  = blockIdx.x * RPB;
    const int i0   = i0b + rblk * 16;

    const int   irow = i0 + (lane & 15);        // A-fragment row for this lane
    const float s1i  = s1[head * NN + irow];
    const float mi   = M[head * NN + irow];

    v8f acc[NT] = {};

    for (int j0 = 0; j0 < NN; j0 += 32) {
#if HAVE_ASYNC
        for (int c = tid; c < RPB * 8; c += 128) {            // 16B chunks of adj
            const int r = c >> 3, q = c & 7;
            async_cp16(&lAdj[r * 32 + q * 4],
                       &adj[(size_t)(i0b + r) * NN + j0 + q * 4]);
        }
        for (int c = tid; c < HEADS * FDIM * 4; c += 128) {   // 16B chunks of hT
            const int row = c >> 2, q = c & 3;
            async_cp16(&lHT[row * KS + q * 8],
                       &HbT[(size_t)row * NN + j0 + q * 8]);
        }
        if (tid < HEADS * 32) lS2[tid] = s2[(tid >> 5) * NN + j0 + (tid & 31)];
        __builtin_amdgcn_s_wait_asynccnt(0);
        __syncthreads();
#else
        for (int c = tid; c < RPB * 8; c += 128) {
            const int r = c >> 3, q = c & 7;
            *(uint4*)&lAdj[r * 32 + q * 4] =
                *(const uint4*)&adj[(size_t)(i0b + r) * NN + j0 + q * 4];
        }
        for (int c = tid; c < HEADS * FDIM * 4; c += 128) {
            const int row = c >> 2, q = c & 3;
            *(uint4*)&lHT[row * KS + q * 8] =
                *(const uint4*)&HbT[(size_t)row * NN + j0 + q * 8];
        }
        if (tid < HEADS * 32) lS2[tid] = s2[(tid >> 5) * NN + j0 + (tid & 31)];
        __syncthreads();
#endif

        Frag16 af;
#pragma unroll
        for (int idx = 0; idx < 16; ++idx) {
            const int   k = a_k(lane, idx);
            const int   a = lAdj[(rblk * 16 + (lane & 15)) * 32 + k];
            const float e = (a > 0) ? lrelu(s1i + lS2[head * 32 + k]) : NEG_INF;
            af.v[idx] = (__bf16)__expf(e - mi);
        }
#pragma unroll
        for (int t = 0; t < NT; ++t) {
            Frag16 bf;
            const __bf16* bp =
                &lHT[(head * FDIM + t * 16 + (lane & 15)) * KS + ((lane >> 4) << 3)];
            bf.u[0] = *(const uint4*)(bp);
            bf.u[1] = *(const uint4*)(bp + 16);
            acc[t] = __builtin_amdgcn_wmma_f32_16x16x32_bf16(
                false, af.v, false, bf.v, (short)0, acc[t], false, false);
        }
        __syncthreads();
    }

#pragma unroll
    for (int t = 0; t < NT; ++t) {
        const int n = t * 16 + (lane & 15);
#pragma unroll
        for (int r = 0; r < 8; ++r) {
            const int   ic = i0 + r + ((lane >> 4) << 3);   // C-fragment row
            const float z  = Z[head * NN + ic];
            Out[((size_t)head * NN + ic) * FDIM + n] = eluf(acc[t][r] / z);
        }
    }
}

// x2 = (sum over 4 heads of elu-outputs) / 4
__global__ void headsum_kernel(const float* __restrict__ Oh, float* __restrict__ X2) {
    int t = blockIdx.x * blockDim.x + threadIdx.x;
    if (t >= NN * 64) return;
    float s = 0.f;
#pragma unroll
    for (int h = 0; h < 4; ++h) s += Oh[(size_t)h * NN * 64 + t];
    X2[t] = 0.25f * s;
}

// ---------------------------------------------------------------------------
extern "C" void kernel_launch(void* const* d_in, const int* in_sizes, int n_in,
                              void* d_out, int out_size, void* d_ws, size_t ws_size,
                              hipStream_t stream) {
    const float* x    = (const float*)d_in[0];
    const int*   adj  = (const int*)d_in[1];
    const float* Wh   = (const float*)d_in[2];
    const float* ah   = (const float*)d_in[3];
    const float* Wo   = (const float*)d_in[4];
    const float* ao   = (const float*)d_in[5];
    float*       out  = (float*)d_out;

    char* p = (char*)d_ws;
    auto alloc = [&](size_t bytes) -> void* {
        void* r = (void*)p;
        p += (bytes + 255) & ~(size_t)255;
        return r;
    };
    float*  h1f = (float*) alloc((size_t)4 * NN * 64 * sizeof(float));
    __bf16* h1b = (__bf16*)alloc((size_t)4 * NN * 64 * sizeof(__bf16));   // [H][F][N]
    float*  s1  = (float*) alloc((size_t)4 * NN * sizeof(float));
    float*  s2  = (float*) alloc((size_t)4 * NN * sizeof(float));
    float*  m1  = (float*) alloc((size_t)4 * NN * sizeof(float));
    float*  z1  = (float*) alloc((size_t)4 * NN * sizeof(float));
    float*  oh  = (float*) alloc((size_t)4 * NN * 64 * sizeof(float));
    float*  x2  = (float*) alloc((size_t)NN * 64 * sizeof(float));
    float*  h2f = (float*) alloc((size_t)NN * 32 * sizeof(float));
    __bf16* h2b = (__bf16*)alloc((size_t)NN * 32 * sizeof(__bf16));       // [F][N]
    float*  s1b = (float*) alloc((size_t)NN * sizeof(float));
    float*  s2b = (float*) alloc((size_t)NN * sizeof(float));
    float*  m2  = (float*) alloc((size_t)NN * sizeof(float));
    float*  z2  = (float*) alloc((size_t)NN * sizeof(float));

    dim3 b128(128);
    // ---- layer 1 (4 heads) ----
    gemm_xw_kernel<256, 64><<<dim3(NN / 64, 4), b128, 0, stream>>>(x, Wh, h1f, h1b);
    s12_kernel<64><<<(4 * NN + 255) / 256, 256, 0, stream>>>(h1f, ah, s1, s2, 4 * NN);
    maxz_kernel<4><<<NN / 8, 256, 0, stream>>>(adj, s1, s2, m1, z1);
    pv_kernel<64, 4><<<NN / 16, b128, 0, stream>>>(adj, s1, s2, m1, z1, h1b, oh);
    headsum_kernel<<<(NN * 64 + 255) / 256, 256, 0, stream>>>(oh, x2);
    // ---- layer 2 (single head, F_out = 32) ----
    gemm_xw_kernel<64, 32><<<dim3(NN / 64, 1), b128, 0, stream>>>(x2, Wo, h2f, h2b);
    s12_kernel<32><<<(NN + 255) / 256, 256, 0, stream>>>(h2f, ao, s1b, s2b, NN);
    maxz_kernel<1><<<NN / 8, 256, 0, stream>>>(adj, s1b, s2b, m2, z2);
    pv_kernel<32, 1><<<NN / 64, b128, 0, stream>>>(adj, s1b, s2b, m2, z2, h2b, out);
}
